// MultiHeadAttention_38156489458328
// MI455X (gfx1250) — matrix operand order repaired and checked by reading, not compiled, not run on hardware
//
#include <hip/hip_runtime.h>
#include <hip/hip_bf16.h>

#define B_  4
#define T_  2048
#define DM_ 1024
#define H_  16
#define DK_ 64
#define DV_ 64

typedef __attribute__((ext_vector_type(16))) __bf16 v16bf;
typedef __attribute__((ext_vector_type(8)))  __bf16 v8bf;
typedef __attribute__((ext_vector_type(4)))  __bf16 v4bf;
typedef __attribute__((ext_vector_type(8)))  float  v8f;
typedef __attribute__((ext_vector_type(4)))  float  v4f;

// A-operand gather: two contiguous 8x-bf16 runs -> v16bf
__device__ __forceinline__ v16bf ld_a16(const __bf16* p0, const __bf16* p1) {
  v8bf lo = *(const v8bf*)p0;
  v8bf hi = *(const v8bf*)p1;
  return __builtin_shufflevector(lo, hi, 0,1,2,3,4,5,6,7,8,9,10,11,12,13,14,15);
}

__device__ __forceinline__ v8f wmma_bf16(v16bf a, v16bf b, v8f c) {
  // D = A(16x32) * B(32x16) + C, fp32 accumulate
  return __builtin_amdgcn_wmma_f32_16x16x32_bf16(false, a, false, b, (short)0, c,
                                                 false, false);
}

// ---------------------------------------------------------------- convert x
__global__ __launch_bounds__(256) void cvt_x_kernel(const float* __restrict__ x,
                                                    __bf16* __restrict__ xb) {
  size_t i = ((size_t)blockIdx.x * 256 + threadIdx.x) * 4;
  v4f v = *(const v4f*)(x + i);
  v4bf o;
  o[0] = (__bf16)v[0]; o[1] = (__bf16)v[1]; o[2] = (__bf16)v[2]; o[3] = (__bf16)v[3];
  *(v4bf*)(xb + i) = o;
}

// ------------------------------------------- transpose weights: Wt[sel][h][n][d]
__global__ __launch_bounds__(256) void cvt_w_kernel(const float* __restrict__ Wq,
                                                    const float* __restrict__ Wk,
                                                    const float* __restrict__ Wv,
                                                    __bf16* __restrict__ Wt) {
  int id = blockIdx.x * 256 + threadIdx.x;     // [sel][h][n][d] flat
  int d    = id & (DM_ - 1);
  int rest = id >> 10;
  int n    = rest & (DK_ - 1);
  rest >>= 6;
  int h    = rest & (H_ - 1);
  int sel  = rest >> 4;
  const float* W = (sel == 0) ? Wq : ((sel == 1) ? Wk : Wv);
  Wt[id] = (__bf16)W[((size_t)h * DM_ + d) * DK_ + n];
}

// ------------------------------------------------------------ QKV projection
// One wave computes a 16(t) x 64(n) tile for one (b,h,sel).
// Q,K stored [b,h,t,dk] bf16 (computed transposed so stores are contiguous);
// V stored transposed [b,h,dv,t] bf16.
__global__ __launch_bounds__(256) void proj_kernel(const __bf16* __restrict__ xb,
                                                   const __bf16* __restrict__ Wt,
                                                   __bf16* __restrict__ Qb,
                                                   __bf16* __restrict__ Kb,
                                                   __bf16* __restrict__ Vt) {
  const int lane = threadIdx.x & 31;
  const int wid  = threadIdx.x >> 5;
  const int gw   = blockIdx.x * 8 + wid;     // 24576 waves total
  const int ttile = gw & 127;                // T_/16
  int tmp = gw >> 7;
  const int bh  = tmp & 63;
  const int sel = tmp >> 6;                  // 0=Q 1=K 2=V
  const int b = bh >> 4, h = bh & 15;
  const int r = lane & 15, hf = lane >> 4;
  const int tbase = ttile * 16;

  const __bf16* wbase = Wt + ((size_t)sel * H_ * DK_ + (size_t)h * DK_) * DM_;
  const __bf16* xrow  = xb + (size_t)(b * T_ + tbase + r) * DM_;
  v8f acc[4] = {};

  if (sel < 2) {
    // D(n,t) = Wt-rows(A: n x d) * x^T(B: d x t)
    for (int kb = 0; kb < DM_; kb += 32) {
      v16bf Bop = *(const v16bf*)(xrow + kb + 16 * hf);
#pragma unroll
      for (int nt = 0; nt < 4; ++nt) {
        const __bf16* wrow = wbase + (size_t)(nt * 16 + r) * DM_;
        v16bf Aop = ld_a16(wrow + kb + 16 * hf, wrow + kb + 16 * hf + 8);
        acc[nt] = wmma_bf16(Aop, Bop, acc[nt]);
      }
    }
    // lane: col t = tbase+r, rows dk = nt*16 + 8*hf + j  (8 contiguous bf16)
    __bf16* dst = (sel == 0 ? Qb : Kb) + (size_t)(bh * T_ + tbase + r) * DK_;
#pragma unroll
    for (int nt = 0; nt < 4; ++nt) {
      v8bf pv;
#pragma unroll
      for (int j = 0; j < 8; ++j) pv[j] = (__bf16)acc[nt][j];
      *(v8bf*)(dst + nt * 16 + 8 * hf) = pv;
    }
  } else {
    // D(t,dv) = x(A: t x d) * Wt(B: d x n)
    for (int kb = 0; kb < DM_; kb += 32) {
      v16bf Aop = ld_a16(xrow + kb + 16 * hf, xrow + kb + 16 * hf + 8);
#pragma unroll
      for (int nt = 0; nt < 4; ++nt) {
        const __bf16* wrow = wbase + (size_t)(nt * 16 + r) * DM_;
        v16bf Bop = *(const v16bf*)(wrow + kb + 16 * hf);
        acc[nt] = wmma_bf16(Aop, Bop, acc[nt]);
      }
    }
    // lane: col dv = nt*16+r, rows t = tbase + 8*hf + j -> Vt[dv][t] contiguous
#pragma unroll
    for (int nt = 0; nt < 4; ++nt) {
      v8bf pv;
#pragma unroll
      for (int j = 0; j < 8; ++j) pv[j] = (__bf16)acc[nt][j];
      __bf16* dst = Vt + (size_t)(bh * DV_ + nt * 16 + r) * T_ + tbase + 8 * hf;
      *(v8bf*)dst = pv;
    }
  }
}

// ------------------------------------------------------------ flash attention
// One wave per (b, h, 16-query tile). S^T = K*Q^T ; O^T = V^T*P^T.
__global__ __launch_bounds__(256) void attn_kernel(const __bf16* __restrict__ Qb,
                                                   const __bf16* __restrict__ Kb,
                                                   const __bf16* __restrict__ Vt,
                                                   float* __restrict__ out) {
  const int lane = threadIdx.x & 31;
  const int wid  = threadIdx.x >> 5;
  const int gw   = blockIdx.x * 8 + wid;     // 8192 waves total
  const int qtile = gw & 127;
  const int bh = gw >> 7;
  const int b = bh >> 4, h = bh & 15;
  const int r = lane & 15, hf = lane >> 4;
  const int qbase = qtile * 16;
  const int mg = qbase + r;                  // this lane's query row
  const float NEG = -__builtin_inff();
  const float scale = 0.125f;                // DK^-0.5

  // Q^T B-operands (col m = r, elements d contiguous), held across key loop
  const __bf16* qrow = Qb + (size_t)(bh * T_ + qbase + r) * DK_;
  v16bf Qop0 = *(const v16bf*)(qrow + 16 * hf);
  v16bf Qop1 = *(const v16bf*)(qrow + 32 + 16 * hf);

  const __bf16* kbase = Kb + (size_t)bh * T_ * DK_;
  const __bf16* vbase = Vt + (size_t)bh * DV_ * T_;

  v8f o[4] = {};
  float M = NEG, L = 0.f;

  const int nsteps = (qbase + 47) >> 5;      // cover keys 0..qbase+15
  for (int it = 0; it < nsteps; ++it) {
    const int sb = it * 32;
    const __bf16* k0 = kbase + (size_t)(sb + r) * DK_;
    const __bf16* k1 = kbase + (size_t)(sb + 16 + r) * DK_;
    if (it + 1 < nsteps)
      __builtin_prefetch(kbase + (size_t)(sb + 32 + r) * DK_, 0, 1);

    v16bf Ka00 = ld_a16(k0 + 16 * hf,      k0 + 16 * hf + 8);
    v16bf Ka01 = ld_a16(k0 + 32 + 16 * hf, k0 + 40 + 16 * hf);
    v16bf Ka10 = ld_a16(k1 + 16 * hf,      k1 + 16 * hf + 8);
    v16bf Ka11 = ld_a16(k1 + 32 + 16 * hf, k1 + 40 + 16 * hf);

    v8f s0 = {}, s1 = {};
    s0 = wmma_bf16(Ka00, Qop0, s0);
    s0 = wmma_bf16(Ka01, Qop1, s0);
    s1 = wmma_bf16(Ka10, Qop0, s1);
    s1 = wmma_bf16(Ka11, Qop1, s1);

    // scale + causal mask; row (= query) max needs only one cross-half shuffle
    float rmax = NEG;
#pragma unroll
    for (int j = 0; j < 8; ++j) {
      int sg0 = sb + 8 * hf + j;
      float v0 = (sg0 <= mg) ? s0[j] * scale : NEG;
      float v1 = (sg0 + 16 <= mg) ? s1[j] * scale : NEG;
      s0[j] = v0; s1[j] = v1;
      rmax = fmaxf(rmax, fmaxf(v0, v1));
    }
    rmax = fmaxf(rmax, __shfl_xor(rmax, 16, 32));
    float Mn = fmaxf(M, rmax);
    float fct = __expf(M - Mn);
    M = Mn;

    float a[8], bb[8], lsum = 0.f;
#pragma unroll
    for (int j = 0; j < 8; ++j) {
      a[j]  = __expf(s0[j] - Mn);            // key s = sb + 8*hf + j
      bb[j] = __expf(s1[j] - Mn);            // key s = sb + 16 + 8*hf + j
      lsum += a[j] + bb[j];
    }
    lsum += __shfl_xor(lsum, 16, 32);
    L = L * fct + lsum;

#pragma unroll
    for (int dt = 0; dt < 4; ++dt) o[dt] *= fct;   // lane-uniform rescale

    // Build P^T B-operand (col m = r, element e <-> s = 16*hf + e)
    v16bf Pb = {};
#pragma unroll
    for (int j = 0; j < 8; ++j) {
      float ax = __shfl_xor(a[j], 16, 32);
      float bx = __shfl_xor(bb[j], 16, 32);
      float lo = hf ? bx : a[j];             // s = 16*hf + j
      float hi = hf ? bb[j] : ax;            // s = 16*hf + 8 + j
      Pb[j]     = (__bf16)lo;
      Pb[j + 8] = (__bf16)hi;
    }

    // O^T += V^T(A: dv x s) * P^T(B: s x m)
#pragma unroll
    for (int dt = 0; dt < 4; ++dt) {
      const __bf16* vrow = vbase + (size_t)(dt * 16 + r) * T_;
      v16bf Va = ld_a16(vrow + sb + 16 * hf, vrow + sb + 16 * hf + 8);
      o[dt] = wmma_bf16(Va, Pb, o[dt]);
    }
  }

  // Normalize and store: O^T lane holds col m = r, rows dv = dt*16 + 8*hf + j
  float inv = 1.0f / L;                      // per-query, uniform across halves
  float* orow = out + (size_t)(b * T_ + qbase + r) * (H_ * DV_) + h * DV_;
#pragma unroll
  for (int dt = 0; dt < 4; ++dt) {
    v4f lo4, hi4;
#pragma unroll
    for (int j = 0; j < 4; ++j) { lo4[j] = o[dt][j] * inv; hi4[j] = o[dt][j + 4] * inv; }
    float* p = orow + dt * 16 + 8 * hf;
    *(v4f*)p = lo4;
    *(v4f*)(p + 4) = hi4;
  }
}

// ---------------------------------------------------------------------------
extern "C" void kernel_launch(void* const* d_in, const int* in_sizes, int n_in,
                              void* d_out, int out_size, void* d_ws, size_t ws_size,
                              hipStream_t stream) {
  const float* x  = (const float*)d_in[0];
  const float* Wq = (const float*)d_in[1];
  const float* Wk = (const float*)d_in[2];
  const float* Wv = (const float*)d_in[3];

  char* ws = (char*)d_ws;
  size_t off = 0;
  __bf16* xb = (__bf16*)(ws + off); off += (size_t)B_ * T_ * DM_ * 2;       // 16 MB
  __bf16* Wt = (__bf16*)(ws + off); off += (size_t)3 * H_ * DK_ * DM_ * 2;  //  6 MB
  __bf16* Qb = (__bf16*)(ws + off); off += (size_t)B_ * H_ * T_ * DK_ * 2;  // 16 MB
  __bf16* Kb = (__bf16*)(ws + off); off += (size_t)B_ * H_ * T_ * DK_ * 2;  // 16 MB
  __bf16* Vt = (__bf16*)(ws + off); off += (size_t)B_ * H_ * DV_ * T_ * 2;  // 16 MB

  cvt_x_kernel<<<(B_ * T_ * DM_) / (256 * 4), 256, 0, stream>>>(x, xb);
  cvt_w_kernel<<<(3 * H_ * DK_ * DM_) / 256, 256, 0, stream>>>(Wq, Wk, Wv, Wt);
  proj_kernel<<<(128 * 64 * 3) / 8, 256, 0, stream>>>(xb, Wt, Qb, Kb, Vt);
  attn_kernel<<<(B_ * H_ * (T_ / 16)) / 8, 256, 0, stream>>>(Qb, Kb, Vt,
                                                             (float*)d_out);
}
